// GATNodeScorer_43800076485316
// MI455X (gfx1250) — matrix operand order repaired and checked by reading, not compiled, not run on hardware
//
#include <hip/hip_runtime.h>
#include <hip/hip_bf16.h>
#include <math.h>

// ---------------------------------------------------------------------------
// Problem constants (match reference)
// ---------------------------------------------------------------------------
#define IN_DIM   1024
#define CODE_DIM 768
#define NUM_DIM  256      // IN_DIM - CODE_DIM
#define HID      256
#define HEADS    4
#define DH       64

typedef __attribute__((ext_vector_type(16))) __bf16 v16bf;
typedef __attribute__((ext_vector_type(8)))  __bf16 v8bf;
typedef __attribute__((ext_vector_type(8)))  float  v8f;
typedef __attribute__((ext_vector_type(4)))  float  f4;

static __device__ inline __bf16 f2bf(float f) {
    unsigned u = __float_as_uint(f);
    u += 0x7FFFu + ((u >> 16) & 1u);           // round-to-nearest-even
    unsigned short s = (unsigned short)(u >> 16);
    __bf16 r;
    __builtin_memcpy(&r, &s, 2);
    return r;
}

static __device__ inline float waveRedSum(float v) {
    #pragma unroll
    for (int off = 16; off > 0; off >>= 1) v += __shfl_down(v, off, 32);
    return v;   // valid in lane 0
}

static __device__ inline void atomicMaxF32(float* addr, float v) {
    if (v >= 0.0f) atomicMax((int*)addr, __float_as_int(v));
    else           atomicMin((unsigned int*)addr, __float_as_uint(v));
}

// ---------------------------------------------------------------------------
// K0: per-column mean / inv-std of x[:, CODE_DIM:]  (256 columns)
// ---------------------------------------------------------------------------
__global__ void k_colstats(const float* __restrict__ x, int N,
                           float* __restrict__ colm, float* __restrict__ colis) {
    __shared__ float ssum[256], ssq[256];
    const int j = blockIdx.x;               // column 0..255
    const int t = threadIdx.x;
    float s = 0.f, q = 0.f;
    for (int i = t; i < N; i += 256) {
        float v = x[(size_t)i * IN_DIM + CODE_DIM + j];
        s += v; q += v * v;
    }
    ssum[t] = s; ssq[t] = q;
    __syncthreads();
    for (int off = 128; off > 0; off >>= 1) {
        if (t < off) { ssum[t] += ssum[t + off]; ssq[t] += ssq[t + off]; }
        __syncthreads();
    }
    if (t == 0) {
        float m = ssum[0] / (float)N;
        float var = (ssq[0] - (float)N * m * m) / (float)(N - 1);
        float sd = sqrtf(fmaxf(var, 0.f));
        colm[j]  = m;
        colis[j] = 1.f / fmaxf(sd, 1e-6f);
    }
}

// ---------------------------------------------------------------------------
// K1: per-row 1/||x[i,0:768]||  (wave per row)
// ---------------------------------------------------------------------------
__global__ void k_rownorm(const float* __restrict__ x, int N,
                          float* __restrict__ nrm_inv) {
    const int wave = threadIdx.x >> 5, lane = threadIdx.x & 31;
    const int i = blockIdx.x * 8 + wave;
    if (i >= N) return;
    const float* row = x + (size_t)i * IN_DIM;
    float s = 0.f;
    for (int j = lane; j < CODE_DIM; j += 32) { float v = row[j]; s += v * v; }
    s = waveRedSum(s);
    if (lane == 0) nrm_inv[i] = 1.f / fmaxf(sqrtf(s), 1e-12f);
}

// ---------------------------------------------------------------------------
// K2: convert weights to N-major bf16:  Wt[n*ID + k] = W[n*ID + k]
// ---------------------------------------------------------------------------
__global__ void k_wt_bf16(const float* __restrict__ W, __bf16* __restrict__ Wt,
                          int ID, int OD) {
    int idx = blockIdx.x * 256 + threadIdx.x;
    if (idx >= ID * OD) return;
    int j = idx / OD, k = idx - j * OD;
    Wt[idx] = f2bf(W[idx]);
}

// ---------------------------------------------------------------------------
// K3: WMMA GEMM  hA = hB = relu(xin @ W_inT + b_in)
//     xin built on the fly from x with fused normalization.
//     Block = 8 waves = 2 row-tiles x 4 col-groups; each wave: 4 col tiles.
//     K loop statically split: k<8 -> numeric region, k>=8 -> code region.
// ---------------------------------------------------------------------------
__global__ void k_gemm_in(const float* __restrict__ x,
                          const float* __restrict__ colm,
                          const float* __restrict__ colis,
                          const float* __restrict__ nrm_inv,
                          const __bf16* __restrict__ Wt,   // [256][1024] bf16
                          const float* __restrict__ bias,
                          float* __restrict__ hA, float* __restrict__ hB, int N) {
    const int wave = threadIdx.x >> 5, lane = threadIdx.x & 31;
    const int half = lane >> 4, m = lane & 15;
    const int rowBase = blockIdx.x * 32 + (wave >> 2) * 16;
    const int cg = (wave & 3) * 64;                 // 4 col tiles: cg..cg+63
    int row = rowBase + m; if (row >= N) row = N - 1;
    const float* xrow = x + (size_t)row * IN_DIM;
    const float  ninv = nrm_inv[row];

    v8f c0 = {}, c1 = {}, c2 = {}, c3 = {};

    // ---- phase 1: K-steps 0..7, xin[j] = (x[768+j]-m[j])*is[j] ----
    #pragma unroll 2
    for (int k = 0; k < 8; ++k) {
        const int kk = k * 32;
        const int j0 = kk + half * 8, j1 = j0 + 16;
        f4 q0 = ((const f4*)(xrow + CODE_DIM + j0))[0];
        f4 q1 = ((const f4*)(xrow + CODE_DIM + j0))[1];
        f4 q2 = ((const f4*)(xrow + CODE_DIM + j1))[0];
        f4 q3 = ((const f4*)(xrow + CODE_DIM + j1))[1];
        v16bf a;
        #pragma unroll
        for (int t = 0; t < 4; ++t) {
            a[t]      = f2bf((q0[t] - colm[j0 + t])     * colis[j0 + t]);
            a[4 + t]  = f2bf((q1[t] - colm[j0 + 4 + t]) * colis[j0 + 4 + t]);
            a[8 + t]  = f2bf((q2[t] - colm[j1 + t])     * colis[j1 + t]);
            a[12 + t] = f2bf((q3[t] - colm[j1 + 4 + t]) * colis[j1 + 4 + t]);
        }
        const __bf16* brow = Wt + (size_t)(cg + m) * IN_DIM + kk + half * 8;
        __builtin_prefetch(brow + 32, 0, 3);
        v16bf b0 = __builtin_shufflevector(*(const v8bf*)(brow), *(const v8bf*)(brow + 16), 0,1,2,3,4,5,6,7,8,9,10,11,12,13,14,15);
        v16bf b1 = __builtin_shufflevector(*(const v8bf*)(brow + 16 * IN_DIM), *(const v8bf*)(brow + 16 * IN_DIM + 16), 0,1,2,3,4,5,6,7,8,9,10,11,12,13,14,15);
        v16bf b2 = __builtin_shufflevector(*(const v8bf*)(brow + 32 * IN_DIM), *(const v8bf*)(brow + 32 * IN_DIM + 16), 0,1,2,3,4,5,6,7,8,9,10,11,12,13,14,15);
        v16bf b3 = __builtin_shufflevector(*(const v8bf*)(brow + 48 * IN_DIM), *(const v8bf*)(brow + 48 * IN_DIM + 16), 0,1,2,3,4,5,6,7,8,9,10,11,12,13,14,15);
        c0 = __builtin_amdgcn_wmma_f32_16x16x32_bf16(false, a, false, b0, (short)0, c0, false, false);
        c1 = __builtin_amdgcn_wmma_f32_16x16x32_bf16(false, a, false, b1, (short)0, c1, false, false);
        c2 = __builtin_amdgcn_wmma_f32_16x16x32_bf16(false, a, false, b2, (short)0, c2, false, false);
        c3 = __builtin_amdgcn_wmma_f32_16x16x32_bf16(false, a, false, b3, (short)0, c3, false, false);
    }

    // ---- phase 2: K-steps 8..31, xin[j] = x[j-256] * ninv ----
    #pragma unroll 2
    for (int k = 8; k < 32; ++k) {
        const int kk = k * 32;
        const int j0 = kk + half * 8 - NUM_DIM, j1 = j0 + 16;
        f4 q0 = ((const f4*)(xrow + j0))[0];
        f4 q1 = ((const f4*)(xrow + j0))[1];
        f4 q2 = ((const f4*)(xrow + j1))[0];
        f4 q3 = ((const f4*)(xrow + j1))[1];
        v16bf a;
        #pragma unroll
        for (int t = 0; t < 4; ++t) {
            a[t]      = f2bf(q0[t] * ninv);
            a[4 + t]  = f2bf(q1[t] * ninv);
            a[8 + t]  = f2bf(q2[t] * ninv);
            a[12 + t] = f2bf(q3[t] * ninv);
        }
        const __bf16* brow = Wt + (size_t)(cg + m) * IN_DIM + kk + half * 8;
        __builtin_prefetch(brow + 32, 0, 3);
        v16bf b0 = __builtin_shufflevector(*(const v8bf*)(brow), *(const v8bf*)(brow + 16), 0,1,2,3,4,5,6,7,8,9,10,11,12,13,14,15);
        v16bf b1 = __builtin_shufflevector(*(const v8bf*)(brow + 16 * IN_DIM), *(const v8bf*)(brow + 16 * IN_DIM + 16), 0,1,2,3,4,5,6,7,8,9,10,11,12,13,14,15);
        v16bf b2 = __builtin_shufflevector(*(const v8bf*)(brow + 32 * IN_DIM), *(const v8bf*)(brow + 32 * IN_DIM + 16), 0,1,2,3,4,5,6,7,8,9,10,11,12,13,14,15);
        v16bf b3 = __builtin_shufflevector(*(const v8bf*)(brow + 48 * IN_DIM), *(const v8bf*)(brow + 48 * IN_DIM + 16), 0,1,2,3,4,5,6,7,8,9,10,11,12,13,14,15);
        c0 = __builtin_amdgcn_wmma_f32_16x16x32_bf16(false, a, false, b0, (short)0, c0, false, false);
        c1 = __builtin_amdgcn_wmma_f32_16x16x32_bf16(false, a, false, b1, (short)0, c1, false, false);
        c2 = __builtin_amdgcn_wmma_f32_16x16x32_bf16(false, a, false, b2, (short)0, c2, false, false);
        c3 = __builtin_amdgcn_wmma_f32_16x16x32_bf16(false, a, false, b3, (short)0, c3, false, false);
    }

    // ---- epilogue: bias + relu, write hA and hB ----
    v8f acc[4] = {c0, c1, c2, c3};
    #pragma unroll
    for (int t = 0; t < 4; ++t) {
        const int col = cg + t * 16 + m;
        const float bv = bias[col];
        #pragma unroll
        for (int v = 0; v < 8; ++v) {
            int rr = rowBase + v + half * 8;
            if (rr >= N) continue;
            float o = fmaxf(acc[t][v] + bv, 0.f);
            size_t idx = (size_t)rr * HID + col;
            hA[idx] = o; hB[idx] = o;
        }
    }
}

// ---------------------------------------------------------------------------
// K4: relational message passing  hB[dst] += hA[src] + rel[type]*w (wave/edge)
// ---------------------------------------------------------------------------
__global__ void k_msg(const float* __restrict__ hA, float* __restrict__ hB,
                      const int* __restrict__ ei, const int* __restrict__ et,
                      const float* __restrict__ ew, const float* __restrict__ rel,
                      int E) {
    const int wave = threadIdx.x >> 5, lane = threadIdx.x & 31;
    const int e = blockIdx.x * 8 + wave;
    if (e >= E) return;
    const int s = ei[e], d = ei[E + e], t = et[e];
    const float w = ew[e];
    const float* hs = hA + (size_t)s * HID + lane * 8;
    const float* rl = rel + (size_t)t * HID + lane * 8;
    float* hd = hB + (size_t)d * HID + lane * 8;
    f4 a0 = ((const f4*)hs)[0], a1 = ((const f4*)hs)[1];
    f4 r0 = ((const f4*)rl)[0], r1 = ((const f4*)rl)[1];
    #pragma unroll
    for (int q = 0; q < 4; ++q) {
        unsafeAtomicAdd(hd + q,     a0[q] + r0[q] * w);
        unsafeAtomicAdd(hd + 4 + q, a1[q] + r1[q] * w);
    }
}

// ---------------------------------------------------------------------------
// K5: f32 -> bf16 copy
// ---------------------------------------------------------------------------
__global__ void k_f32_to_bf16(const float* __restrict__ src, __bf16* __restrict__ dst,
                              int n) {
    int i = blockIdx.x * 256 + threadIdx.x;
    if (i < n) dst[i] = f2bf(src[i]);
}

// ---------------------------------------------------------------------------
// K6: per-layer init: outacc = 0, mx = -inf, den = 0
// ---------------------------------------------------------------------------
__global__ void k_init(float* __restrict__ outacc, float* __restrict__ mx,
                       float* __restrict__ den, int N) {
    int i = blockIdx.x * 256 + threadIdx.x;
    if (i < N * HID) outacc[i] = 0.f;
    if (i < N * HEADS) { ((int*)mx)[i] = 0xFF800000; den[i] = 0.f; }
}

// ---------------------------------------------------------------------------
// K7: WMMA GEMM  xh = h_bf16 @ WtT   (256x256 weights, no bias/activation)
//     Same 2 row-tile x 4 col-group block tiling as k_gemm_in.
// ---------------------------------------------------------------------------
__global__ void k_gemm_h(const __bf16* __restrict__ hb,
                         const __bf16* __restrict__ Wt,   // [256][256] bf16
                         float* __restrict__ xh, int N) {
    const int wave = threadIdx.x >> 5, lane = threadIdx.x & 31;
    const int half = lane >> 4, m = lane & 15;
    const int rowBase = blockIdx.x * 32 + (wave >> 2) * 16;
    const int cg = (wave & 3) * 64;
    int row = rowBase + m; if (row >= N) row = N - 1;
    const __bf16* hrow = hb + (size_t)row * HID;

    v8f c0 = {}, c1 = {}, c2 = {}, c3 = {};
    #pragma unroll
    for (int k = 0; k < HID / 32; ++k) {
        const int kk = k * 32;
        v8bf r0 = *(const v8bf*)(hrow + kk + half * 8);
        v8bf r1 = *(const v8bf*)(hrow + kk + 16 + half * 8);
        v16bf a;
        #pragma unroll
        for (int t = 0; t < 8; ++t) { a[t] = r0[t]; a[8 + t] = r1[t]; }

        const __bf16* brow = Wt + (size_t)(cg + m) * HID + kk + half * 8;
        v16bf b0 = __builtin_shufflevector(*(const v8bf*)(brow), *(const v8bf*)(brow + 16), 0,1,2,3,4,5,6,7,8,9,10,11,12,13,14,15);
        v16bf b1 = __builtin_shufflevector(*(const v8bf*)(brow + 16 * HID), *(const v8bf*)(brow + 16 * HID + 16), 0,1,2,3,4,5,6,7,8,9,10,11,12,13,14,15);
        v16bf b2 = __builtin_shufflevector(*(const v8bf*)(brow + 32 * HID), *(const v8bf*)(brow + 32 * HID + 16), 0,1,2,3,4,5,6,7,8,9,10,11,12,13,14,15);
        v16bf b3 = __builtin_shufflevector(*(const v8bf*)(brow + 48 * HID), *(const v8bf*)(brow + 48 * HID + 16), 0,1,2,3,4,5,6,7,8,9,10,11,12,13,14,15);
        c0 = __builtin_amdgcn_wmma_f32_16x16x32_bf16(false, a, false, b0, (short)0, c0, false, false);
        c1 = __builtin_amdgcn_wmma_f32_16x16x32_bf16(false, a, false, b1, (short)0, c1, false, false);
        c2 = __builtin_amdgcn_wmma_f32_16x16x32_bf16(false, a, false, b2, (short)0, c2, false, false);
        c3 = __builtin_amdgcn_wmma_f32_16x16x32_bf16(false, a, false, b3, (short)0, c3, false, false);
    }
    v8f acc[4] = {c0, c1, c2, c3};
    #pragma unroll
    for (int t = 0; t < 4; ++t) {
        const int col = cg + t * 16 + m;
        #pragma unroll
        for (int v = 0; v < 8; ++v) {
            int rr = rowBase + v + half * 8;
            if (rr >= N) continue;
            xh[(size_t)rr * HID + col] = acc[t][v];
        }
    }
}

// ---------------------------------------------------------------------------
// K8: attention logits  al_s[i,h] = <xh[i,h,:], a_src[h,:]>, same for dst
// ---------------------------------------------------------------------------
__global__ void k_al(const float* __restrict__ xh,
                     const float* __restrict__ a_src, const float* __restrict__ a_dst,
                     float* __restrict__ al_s, float* __restrict__ al_d, int N) {
    const int wave = threadIdx.x >> 5, lane = threadIdx.x & 31;
    const int i = blockIdx.x * 8 + wave;
    if (i >= N) return;
    const float* r = xh + (size_t)i * HID + lane * 8;
    const float* as = a_src + lane * 8;   // HEADS*DH == HID contiguous
    const float* ad = a_dst + lane * 8;
    float ps = 0.f, pd = 0.f;
    #pragma unroll
    for (int t = 0; t < 8; ++t) { float v = r[t]; ps += v * as[t]; pd += v * ad[t]; }
    #pragma unroll
    for (int off = 4; off > 0; off >>= 1) {
        ps += __shfl_down(ps, off, 8);
        pd += __shfl_down(pd, off, 8);
    }
    if ((lane & 7) == 0) {
        int h = lane >> 3;
        al_s[i * HEADS + h] = ps;
        al_d[i * HEADS + h] = pd;
    }
}

// ---------------------------------------------------------------------------
// K9: edge pass 1 — e = leaky_relu(al_s[s]+al_d[d]); segment max over dst
// ---------------------------------------------------------------------------
__global__ void k_edge1(const int* __restrict__ ei,
                        const float* __restrict__ al_s, const float* __restrict__ al_d,
                        float* __restrict__ ebuf, float* __restrict__ mx,
                        int E, int EN) {
    int tid = blockIdx.x * 256 + threadIdx.x;
    int ed = tid >> 2, h = tid & 3;
    if (ed >= EN) return;
    int s, d;
    if (ed < E) { s = ei[ed]; d = ei[E + ed]; } else { s = d = ed - E; }
    float e = al_s[s * HEADS + h] + al_d[d * HEADS + h];
    e = (e > 0.f) ? e : 0.2f * e;
    ebuf[ed * HEADS + h] = e;
    atomicMaxF32(mx + d * HEADS + h, e);
}

// ---------------------------------------------------------------------------
// K10: edge pass 2 — ex = exp(e - mx[d]); den[d] += ex
// ---------------------------------------------------------------------------
__global__ void k_edge2(const int* __restrict__ ei,
                        float* __restrict__ ebuf, const float* __restrict__ mx,
                        float* __restrict__ den, int E, int EN) {
    int tid = blockIdx.x * 256 + threadIdx.x;
    int ed = tid >> 2, h = tid & 3;
    if (ed >= EN) return;
    int d = (ed < E) ? ei[E + ed] : (ed - E);
    float ex = __expf(ebuf[ed * HEADS + h] - mx[d * HEADS + h]);
    ebuf[ed * HEADS + h] = ex;
    unsafeAtomicAdd(den + d * HEADS + h, ex);
}

// ---------------------------------------------------------------------------
// K11: edge pass 3 — outacc[d] += xh[s] * (ex / (den[d]+eps))   (wave/edge)
// ---------------------------------------------------------------------------
__global__ void k_edge3(const int* __restrict__ ei,
                        const float* __restrict__ ebuf, const float* __restrict__ den,
                        const float* __restrict__ xh, float* __restrict__ outacc,
                        int E, int EN) {
    const int wave = threadIdx.x >> 5, lane = threadIdx.x & 31;
    const int ed = blockIdx.x * 8 + wave;
    if (ed >= EN) return;
    int s, d;
    if (ed < E) { s = ei[ed]; d = ei[E + ed]; } else { s = d = ed - E; }
    const int h = lane >> 3;   // 8 lanes per head, 8 dims per lane
    float alpha = ebuf[ed * HEADS + h] / (den[d * HEADS + h] + 1e-16f);
    const float* xs = xh + (size_t)s * HID + lane * 8;
    float* od = outacc + (size_t)d * HID + lane * 8;
    f4 a0 = ((const f4*)xs)[0], a1 = ((const f4*)xs)[1];
    #pragma unroll
    for (int q = 0; q < 4; ++q) {
        unsafeAtomicAdd(od + q,     a0[q] * alpha);
        unsafeAtomicAdd(od + 4 + q, a1[q] * alpha);
    }
}

// ---------------------------------------------------------------------------
// K12: residual + bias + LayerNorm   dst = LN(outacc + bias + resid)*g + b
// ---------------------------------------------------------------------------
__global__ void k_ln(const float* __restrict__ outacc, const float* __restrict__ bias,
                     const float* __restrict__ resid,
                     const float* __restrict__ g, const float* __restrict__ b,
                     float* __restrict__ dst, int N) {
    const int wave = threadIdx.x >> 5, lane = threadIdx.x & 31;
    const int i = blockIdx.x * 8 + wave;
    if (i >= N) return;
    const float* oa = outacc + (size_t)i * HID + lane * 8;
    const float* rs = resid  + (size_t)i * HID + lane * 8;
    const float* bs = bias + lane * 8;
    float val[8]; float s = 0.f, q = 0.f;
    #pragma unroll
    for (int t = 0; t < 8; ++t) {
        float v = oa[t] + bs[t] + rs[t];
        val[t] = v; s += v; q += v * v;
    }
    s = waveRedSum(s); q = waveRedSum(q);
    s = __shfl(s, 0, 32); q = __shfl(q, 0, 32);
    float mean = s * (1.f / HID);
    float var  = q * (1.f / HID) - mean * mean;
    float rstd = rsqrtf(var + 1e-5f);
    float* out = dst + (size_t)i * HID + lane * 8;
    const float* gg = g + lane * 8; const float* bb = b + lane * 8;
    #pragma unroll
    for (int t = 0; t < 8; ++t) out[t] = (val[t] - mean) * rstd * gg[t] + bb[t];
}

// ---------------------------------------------------------------------------
// K13: final score  out[i] = <h[i,:], W_out> + b_out
// ---------------------------------------------------------------------------
__global__ void k_out(const float* __restrict__ h, const float* __restrict__ Wout,
                      const float* __restrict__ bout, float* __restrict__ out, int N) {
    const int wave = threadIdx.x >> 5, lane = threadIdx.x & 31;
    const int i = blockIdx.x * 8 + wave;
    if (i >= N) return;
    const float* r = h + (size_t)i * HID + lane * 8;
    const float* w = Wout + lane * 8;
    float s = 0.f;
    #pragma unroll
    for (int t = 0; t < 8; ++t) s += r[t] * w[t];
    s = waveRedSum(s);
    if (lane == 0) out[i] = s + bout[0];
}

// ---------------------------------------------------------------------------
// Host launcher
// ---------------------------------------------------------------------------
extern "C" void kernel_launch(void* const* d_in, const int* in_sizes, int n_in,
                              void* d_out, int out_size, void* d_ws, size_t ws_size,
                              hipStream_t stream) {
    const float* x    = (const float*)d_in[0];
    const int*   ei   = (const int*)  d_in[1];
    const int*   et   = (const int*)  d_in[2];
    const float* ew   = (const float*)d_in[3];
    const float* rel  = (const float*)d_in[4];
    const float* W_in = (const float*)d_in[5];
    const float* b_in = (const float*)d_in[6];
    const float* W1   = (const float*)d_in[7];
    const float* a1s  = (const float*)d_in[8];
    const float* a1d  = (const float*)d_in[9];
    const float* b1   = (const float*)d_in[10];
    const float* W2   = (const float*)d_in[11];
    const float* a2s  = (const float*)d_in[12];
    const float* a2d  = (const float*)d_in[13];
    const float* b2   = (const float*)d_in[14];
    const float* ln1g = (const float*)d_in[15];
    const float* ln1b = (const float*)d_in[16];
    const float* ln2g = (const float*)d_in[17];
    const float* ln2b = (const float*)d_in[18];
    const float* Wout = (const float*)d_in[19];
    const float* bout = (const float*)d_in[20];

    const int N  = in_sizes[0] / IN_DIM;
    const int E  = in_sizes[3];
    const int EN = E + N;

    // --- workspace carve-up (256B-aligned) ---
    char* p = (char*)d_ws;
    auto alloc = [&](size_t bytes) {
        void* r = p;
        p += (bytes + 255) & ~((size_t)255);
        return r;
    };
    float*  hA      = (float*) alloc((size_t)N * HID * 4);   // reused as outacc
    float*  hB      = (float*) alloc((size_t)N * HID * 4);   // reused as final h
    float*  hC      = (float*) alloc((size_t)N * HID * 4);
    float*  xh      = (float*) alloc((size_t)N * HID * 4);
    __bf16* hbf     = (__bf16*)alloc((size_t)N * HID * 2);
    __bf16* WtIn    = (__bf16*)alloc((size_t)IN_DIM * HID * 2);
    __bf16* Wt1     = (__bf16*)alloc((size_t)HID * HID * 2);
    __bf16* Wt2     = (__bf16*)alloc((size_t)HID * HID * 2);
    float*  colm    = (float*) alloc(NUM_DIM * 4);
    float*  colis   = (float*) alloc(NUM_DIM * 4);
    float*  nrm_inv = (float*) alloc((size_t)N * 4);
    float*  al_s    = (float*) alloc((size_t)N * HEADS * 4);
    float*  al_d    = (float*) alloc((size_t)N * HEADS * 4);
    float*  mx      = (float*) alloc((size_t)N * HEADS * 4);
    float*  den     = (float*) alloc((size_t)N * HEADS * 4);
    float*  ebuf    = (float*) alloc((size_t)EN * HEADS * 4);
    float*  outacc  = hA;   // hA dead after k_msg

    const dim3 B256(256);
    const int wavesN  = (N + 7) / 8;
    const int wavesE  = (E + 7) / 8;
    const int wavesEN = (EN + 7) / 8;
    const int tiles32 = (N + 31) / 32;

    // ---- preprocessing ----
    k_colstats<<<dim3(NUM_DIM), B256, 0, stream>>>(x, N, colm, colis);
    k_rownorm <<<dim3(wavesN),  B256, 0, stream>>>(x, N, nrm_inv);
    k_wt_bf16 <<<dim3((IN_DIM * HID + 255) / 256), B256, 0, stream>>>(W_in, WtIn, IN_DIM, HID);
    k_wt_bf16 <<<dim3((HID * HID + 255) / 256),    B256, 0, stream>>>(W1, Wt1, HID, HID);
    k_wt_bf16 <<<dim3((HID * HID + 255) / 256),    B256, 0, stream>>>(W2, Wt2, HID, HID);

    // ---- input GEMM (WMMA) + relational scatter ----
    k_gemm_in<<<dim3(tiles32), B256, 0, stream>>>(x, colm, colis, nrm_inv, WtIn, b_in, hA, hB, N);
    k_msg    <<<dim3(wavesE),  B256, 0, stream>>>(hA, hB, ei, et, ew, rel, E);

    const int gNH  = (N * HID + 255) / 256;
    const int gENH = (EN * HEADS + 255) / 256;

    // ---- GAT layer 1:  hB -> hC ----
    k_f32_to_bf16<<<dim3(gNH), B256, 0, stream>>>(hB, hbf, N * HID);
    k_init <<<dim3(gNH),     B256, 0, stream>>>(outacc, mx, den, N);
    k_gemm_h<<<dim3(tiles32), B256, 0, stream>>>(hbf, Wt1, xh, N);
    k_al   <<<dim3(wavesN),  B256, 0, stream>>>(xh, a1s, a1d, al_s, al_d, N);
    k_edge1<<<dim3(gENH),    B256, 0, stream>>>(ei, al_s, al_d, ebuf, mx, E, EN);
    k_edge2<<<dim3(gENH),    B256, 0, stream>>>(ei, ebuf, mx, den, E, EN);
    k_edge3<<<dim3(wavesEN), B256, 0, stream>>>(ei, ebuf, den, xh, outacc, E, EN);
    k_ln   <<<dim3(wavesN),  B256, 0, stream>>>(outacc, b1, hB, ln1g, ln1b, hC, N);

    // ---- GAT layer 2:  hC -> hB (reuse) ----
    k_f32_to_bf16<<<dim3(gNH), B256, 0, stream>>>(hC, hbf, N * HID);
    k_init <<<dim3(gNH),     B256, 0, stream>>>(outacc, mx, den, N);
    k_gemm_h<<<dim3(tiles32), B256, 0, stream>>>(hbf, Wt2, xh, N);
    k_al   <<<dim3(wavesN),  B256, 0, stream>>>(xh, a2s, a2d, al_s, al_d, N);
    k_edge1<<<dim3(gENH),    B256, 0, stream>>>(ei, al_s, al_d, ebuf, mx, E, EN);
    k_edge2<<<dim3(gENH),    B256, 0, stream>>>(ei, ebuf, mx, den, E, EN);
    k_edge3<<<dim3(wavesEN), B256, 0, stream>>>(ei, ebuf, den, xh, outacc, E, EN);
    k_ln   <<<dim3(wavesN),  B256, 0, stream>>>(outacc, b2, hC, ln2g, ln2b, hB, N);

    // ---- final score ----
    k_out<<<dim3(wavesN), B256, 0, stream>>>(hB, Wout, bout, (float*)d_out, N);
}
